// Reduction_Block_71451075936632
// MI455X (gfx1250) — 8-bit matrix operand order repaired and checked by reading, compiled, not run on hardware
//
#include <hip/hip_runtime.h>

typedef __attribute__((ext_vector_type(8))) int v8i;
typedef int vint4 __attribute__((vector_size(16)));

#define BATCH   64
#define CCH     256
#define HIN     28
#define HOUT    14
#define NPIX    (BATCH*HOUT*HOUT)
#define NPIX_IN (BATCH*HIN*HIN)
#define K1      2304
#define KB1     36
#define KB2     4
#define NT1     16
#define NT2     32

#if defined(__HIP_DEVICE_COMPILE__) && __has_builtin(__builtin_amdgcn_global_load_async_to_lds_b128)
#define HAVE_ASYNC_LDS 1
__device__ __forceinline__ void async_g2l_b128(const void* g, void* l) {
    __builtin_amdgcn_global_load_async_to_lds_b128(
        (vint4*)(unsigned long long)g,
        (vint4*)l,
        0, 0);
}
#else
#define HAVE_ASYNC_LDS 0
#endif

#if defined(__HIP_DEVICE_COMPILE__) && __has_builtin(__builtin_amdgcn_s_wait_asynccnt)
#define ASYNC_WAIT(n) __builtin_amdgcn_s_wait_asynccnt(n)
#elif defined(__HIP_DEVICE_COMPILE__)
#define ASYNC_WAIT(n) asm volatile("s_wait_asynccnt %0" :: "i"(n) : "memory")
#else
#define ASYNC_WAIT(n)
#endif

__device__ __forceinline__ signed char sgn8(float v) {
    return v > 0.f ? (signed char)1 : (v < 0.f ? (signed char)-1 : (signed char)0);
}

__global__ __launch_bounds__(256)
void prep_w3x3(const float* __restrict__ w, const float* __restrict__ g,
               const float* __restrict__ bb, const float* __restrict__ mm,
               const float* __restrict__ vv,
               int* __restrict__ Bpack, float* __restrict__ chanA, float* __restrict__ chanB)
{
    const int o = blockIdx.x, tid = threadIdx.x;
    __shared__ float red[256];
    float s = 0.f;
    for (int idx = tid; idx < K1; idx += 256) s += fabsf(w[o * K1 + idx]);
    red[tid] = s; __syncthreads();
    for (int off = 128; off > 0; off >>= 1) {
        if (tid < off) red[tid] += red[tid + off];
        __syncthreads();
    }
    if (tid == 0) {
        float scale = red[0] * (1.f / 2304.f);
        float inv = g[o] * rsqrtf(vv[o] + 1e-5f);
        chanA[o] = scale * inv;
        chanB[o] = bb[o] - mm[o] * inv;
    }
    const int nt = o >> 4, ln = o & 15;
    for (int idx = tid; idx < K1 / 4; idx += 256) {
        const int kb = idx >> 4, rem = idx & 15, half = rem >> 3, j = rem & 7;
        unsigned int dw = 0;
        for (int byte = 0; byte < 4; ++byte) {
            int kk = (j >> 1) * 16 + half * 8 + (j & 1) * 4 + byte;
            int k = kb * 64 + kk;
            int ci = k & 255, tap = k >> 8;
            float wv = w[((o * CCH + ci) * 3 + tap / 3) * 3 + (tap % 3)];
            dw |= ((unsigned int)(unsigned char)sgn8(wv)) << (byte * 8);
        }
        Bpack[(((size_t)kb * NT1 + nt) * 32 + (half * 16 + ln)) * 8 + j] = dw;
    }
}

__global__ __launch_bounds__(256)
void prep_w1x1(const float* __restrict__ w, const float* __restrict__ g,
               const float* __restrict__ bb, const float* __restrict__ mm,
               const float* __restrict__ vv,
               int* __restrict__ Bpack, float* __restrict__ chanA, float* __restrict__ chanB,
               int hsel)
{
    const int o = blockIdx.x, tid = threadIdx.x;
    __shared__ float red[256];
    red[tid] = fabsf(w[o * CCH + tid]); __syncthreads();
    for (int off = 128; off > 0; off >>= 1) {
        if (tid < off) red[tid] += red[tid + off];
        __syncthreads();
    }
    if (tid == 0) {
        float scale = red[0] * (1.f / 256.f);
        float inv = g[o] * rsqrtf(vv[o] + 1e-5f);
        chanA[hsel * CCH + o] = scale * inv;
        chanB[hsel * CCH + o] = bb[o] - mm[o] * inv;
    }
    const int nt = (o >> 4) + hsel * NT1, ln = o & 15;
    if (tid < 64) {
        const int kb = tid >> 4, rem = tid & 15, half = rem >> 3, j = rem & 7;
        unsigned int dw = 0;
        for (int byte = 0; byte < 4; ++byte) {
            int kk = (j >> 1) * 16 + half * 8 + (j & 1) * 4 + byte;
            float wv = w[o * CCH + kb * 64 + kk];
            dw |= ((unsigned int)(unsigned char)sgn8(wv)) << (byte * 8);
        }
        Bpack[(((size_t)kb * NT2 + nt) * 32 + (half * 16 + ln)) * 8 + j] = dw;
    }
}

__global__ __launch_bounds__(256)
void binarize_x(const float* __restrict__ x, const float* __restrict__ rsb,
                signed char* __restrict__ s1)
{
    const int pix = blockIdx.x * blockDim.x + threadIdx.x;
    if (pix >= NPIX_IN) return;
    const int b = pix / (HIN * HIN), r = pix % (HIN * HIN);
    const float* xp = x + (size_t)b * CCH * HIN * HIN + r;
    signed char* sp = s1 + (size_t)pix * CCH;
    for (int c = 0; c < CCH; ++c) {
        float v = xp[(size_t)c * (HIN * HIN)] + rsb[c];
        sp[c] = sgn8(v);
    }
}

__global__ __launch_bounds__(256)
void avgpool_k(const float* __restrict__ x, float* __restrict__ avgp)
{
    const int idx = blockIdx.x * blockDim.x + threadIdx.x;
    if (idx >= BATCH * CCH * HOUT * HOUT) return;
    const int ow = idx % HOUT, oh = (idx / HOUT) % HOUT;
    const int c = (idx / (HOUT * HOUT)) % CCH, b = idx / (HOUT * HOUT * CCH);
    const float* xp = x + (((size_t)(b * CCH + c) * HIN + oh * 2) * HIN + ow * 2);
    float v = 0.25f * (xp[0] + xp[1] + xp[HIN] + xp[HIN + 1]);
    avgp[((size_t)(b * HOUT * HOUT) + oh * HOUT + ow) * CCH + c] = v;
}

__global__ __launch_bounds__(256)
void gemm1(const signed char* __restrict__ s1, const int* __restrict__ Bpack,
           const float* __restrict__ chanA, const float* __restrict__ chanB,
           const float* __restrict__ avgp,
           const float* __restrict__ prg, const float* __restrict__ prb,
           const float* __restrict__ prz, const float* __restrict__ rsb,
           const signed char* __restrict__ zbuf,
           float* __restrict__ out2, signed char* __restrict__ s2)
{
    __shared__ signed char sA[2][128 * 64];
    const int tid = threadIdx.x, lane = tid & 31, wave = tid >> 5;
    const int mw = wave & 3, nw = wave >> 2;
    const int mblock = blockIdx.x, nblock = blockIdx.y;
    const int half = lane >> 4, lm = lane & 15;

    const int arow = tid >> 1, ahalf = tid & 1;
    const int gm_a = mblock * 128 + arow;
    const int ba = gm_a / (HOUT * HOUT), pa = gm_a % (HOUT * HOUT);
    const int oh = pa / HOUT, ow = pa % HOUT;

    v8i acc[2][2];
    acc[0][0] = (v8i)0; acc[0][1] = (v8i)0; acc[1][0] = (v8i)0; acc[1][1] = (v8i)0;

#if HAVE_ASYNC_LDS
    auto issueA = [&](int kb) {
        const int tap = kb >> 2, c0 = (kb & 3) << 6;
        const int ih = oh * 2 - 1 + tap / 3, iw = ow * 2 - 1 + tap % 3;
        const signed char* g = zbuf + ahalf * 32;
        if (ih >= 0 && ih < HIN && iw >= 0 && iw < HIN)
            g = s1 + ((((size_t)ba * HIN + ih) * HIN + iw) << 8) + c0 + ahalf * 32;
        signed char* l = &sA[kb & 1][arow * 64 + ahalf * 32];
        async_g2l_b128(g, l);
        async_g2l_b128(g + 16, l + 16);
    };
    issueA(0);
#endif

    for (int kb = 0; kb < KB1; ++kb) {
#if HAVE_ASYNC_LDS
        if (kb + 1 < KB1) { issueA(kb + 1); ASYNC_WAIT(2); }
        else              { ASYNC_WAIT(0); }
        __syncthreads();
        const signed char* sAc = &sA[kb & 1][0];
#else
        __syncthreads();
        {
            const int tap = kb >> 2, c0 = (kb & 3) << 6;
            const int ih = oh * 2 - 1 + tap / 3, iw = ow * 2 - 1 + tap % 3;
            uint4 v0 = make_uint4(0, 0, 0, 0), v1 = v0;
            if (ih >= 0 && ih < HIN && iw >= 0 && iw < HIN) {
                const uint4* src = (const uint4*)(s1 +
                    ((((size_t)ba * HIN + ih) * HIN + iw) << 8) + c0 + ahalf * 32);
                v0 = src[0]; v1 = src[1];
            }
            uint4* dst = (uint4*)&sA[0][arow * 64 + ahalf * 32];
            dst[0] = v0; dst[1] = v1;
        }
        __syncthreads();
        const signed char* sAc = &sA[0][0];
#endif

        v8i afrag[2], bfrag[2];
        #pragma unroll
        for (int mi = 0; mi < 2; ++mi) {
            const unsigned int* rp = (const unsigned int*)&sAc[(mw * 32 + mi * 16 + lm) * 64];
            const int bse = half * 2;
            v8i a;
            a[0] = rp[bse + 0];  a[1] = rp[bse + 1];
            a[2] = rp[bse + 4];  a[3] = rp[bse + 5];
            a[4] = rp[bse + 8];  a[5] = rp[bse + 9];
            a[6] = rp[bse + 12]; a[7] = rp[bse + 13];
            afrag[mi] = a;
        }
        #pragma unroll
        for (int ni = 0; ni < 2; ++ni) {
            const int nt = (nblock * 64 + nw * 32 + ni * 16) >> 4;
            bfrag[ni] = *(const v8i*)(Bpack + (((size_t)kb * NT1 + nt) * 32 + lane) * 8);
        }
        if (kb + 1 < KB1) {
            const int nt = (nblock * 64 + nw * 32) >> 4;
            __builtin_prefetch(Bpack + (((size_t)(kb + 1) * NT1 + nt) * 32 + lane) * 8, 0, 1);
        }
        #pragma unroll
        for (int mi = 0; mi < 2; ++mi)
            #pragma unroll
            for (int ni = 0; ni < 2; ++ni)
                acc[mi][ni] = __builtin_amdgcn_wmma_i32_16x16x64_iu8(
                    true, afrag[mi], true, bfrag[ni], acc[mi][ni], false, false);
        __syncthreads();
    }

    #pragma unroll
    for (int mi = 0; mi < 2; ++mi) {
        #pragma unroll
        for (int ni = 0; ni < 2; ++ni) {
            const int n0 = nblock * 64 + nw * 32 + ni * 16 + lm;
            const float cA = chanA[n0], cB = chanB[n0];
            const float g = prg[n0], bt = prb[n0], z = prz[n0], rb = rsb[n0];
            #pragma unroll
            for (int r = 0; r < 8; ++r) {
                const int gm = mblock * 128 + mw * 32 + mi * 16 + r + half * 8;
                float v = (float)acc[mi][ni][r] * cA + cB + avgp[(size_t)gm * CCH + n0];
                float t = v - g;
                float o2 = (t > 0.f ? t : bt * t) + z;
                out2[(size_t)gm * CCH + n0] = o2;
                s2[(size_t)gm * CCH + n0] = sgn8(o2 + rb);
            }
        }
    }
}

__global__ __launch_bounds__(256)
void gemm2(const signed char* __restrict__ s2, const int* __restrict__ Bpack,
           const float* __restrict__ chanA, const float* __restrict__ chanB,
           const float* __restrict__ out2,
           const float* __restrict__ prg, const float* __restrict__ prb,
           const float* __restrict__ prz,
           float* __restrict__ outp)
{
    __shared__ signed char sA[2][128 * 64];
    const int tid = threadIdx.x, lane = tid & 31, wave = tid >> 5;
    const int mw = wave & 3, nw = wave >> 2;
    const int mblock = blockIdx.x, nblock = blockIdx.y;
    const int half = lane >> 4, lm = lane & 15;

    const int arow = tid >> 1, ahalf = tid & 1;
    const int gm_a = mblock * 128 + arow;

    v8i acc[2][2];
    acc[0][0] = (v8i)0; acc[0][1] = (v8i)0; acc[1][0] = (v8i)0; acc[1][1] = (v8i)0;

#if HAVE_ASYNC_LDS
    auto issueA = [&](int kb) {
        const signed char* g = s2 + ((size_t)gm_a << 8) + (kb << 6) + ahalf * 32;
        signed char* l = &sA[kb & 1][arow * 64 + ahalf * 32];
        async_g2l_b128(g, l);
        async_g2l_b128(g + 16, l + 16);
    };
    issueA(0);
#endif

    for (int kb = 0; kb < KB2; ++kb) {
#if HAVE_ASYNC_LDS
        if (kb + 1 < KB2) { issueA(kb + 1); ASYNC_WAIT(2); }
        else              { ASYNC_WAIT(0); }
        __syncthreads();
        const signed char* sAc = &sA[kb & 1][0];
#else
        __syncthreads();
        {
            const uint4* src = (const uint4*)(s2 + ((size_t)gm_a << 8) + (kb << 6) + ahalf * 32);
            uint4* dst = (uint4*)&sA[0][arow * 64 + ahalf * 32];
            dst[0] = src[0]; dst[1] = src[1];
        }
        __syncthreads();
        const signed char* sAc = &sA[0][0];
#endif

        v8i afrag[2], bfrag[2];
        #pragma unroll
        for (int mi = 0; mi < 2; ++mi) {
            const unsigned int* rp = (const unsigned int*)&sAc[(mw * 32 + mi * 16 + lm) * 64];
            const int bse = half * 2;
            v8i a;
            a[0] = rp[bse + 0];  a[1] = rp[bse + 1];
            a[2] = rp[bse + 4];  a[3] = rp[bse + 5];
            a[4] = rp[bse + 8];  a[5] = rp[bse + 9];
            a[6] = rp[bse + 12]; a[7] = rp[bse + 13];
            afrag[mi] = a;
        }
        #pragma unroll
        for (int ni = 0; ni < 2; ++ni) {
            const int nt = (nblock * 64 + nw * 32 + ni * 16) >> 4;
            bfrag[ni] = *(const v8i*)(Bpack + (((size_t)kb * NT2 + nt) * 32 + lane) * 8);
        }
        if (kb + 1 < KB2) {
            const int nt = (nblock * 64 + nw * 32) >> 4;
            __builtin_prefetch(Bpack + (((size_t)(kb + 1) * NT2 + nt) * 32 + lane) * 8, 0, 1);
        }
        #pragma unroll
        for (int mi = 0; mi < 2; ++mi)
            #pragma unroll
            for (int ni = 0; ni < 2; ++ni)
                acc[mi][ni] = __builtin_amdgcn_wmma_i32_16x16x64_iu8(
                    true, afrag[mi], true, bfrag[ni], acc[mi][ni], false, false);
        __syncthreads();
    }

    #pragma unroll
    for (int mi = 0; mi < 2; ++mi) {
        #pragma unroll
        for (int ni = 0; ni < 2; ++ni) {
            const int n = nblock * 64 + nw * 32 + ni * 16 + lm;
            const int nc = n & 255;
            const float cA = chanA[n], cB = chanB[n];
            const float g = prg[nc], bt = prb[nc], z = prz[nc];
            #pragma unroll
            for (int r = 0; r < 8; ++r) {
                const int gm = mblock * 128 + mw * 32 + mi * 16 + r + half * 8;
                float v = (float)acc[mi][ni][r] * cA + cB + out2[(size_t)gm * CCH + nc];
                float t = v - g;
                float res = (t > 0.f ? t : bt * t) + z;
                const int bpix = gm / (HOUT * HOUT), p = gm % (HOUT * HOUT);
                outp[((size_t)bpix * 512 + n) * (HOUT * HOUT) + p] = res;
            }
        }
    }
}

extern "C" void kernel_launch(void* const* d_in, const int* in_sizes, int n_in,
                              void* d_out, int out_size, void* d_ws, size_t ws_size,
                              hipStream_t stream)
{
    const float* x     = (const float*)d_in[0];
    const float* rsb   = (const float*)d_in[1];
    const float* w1    = (const float*)d_in[2];
    const float* bn1g  = (const float*)d_in[3];
    const float* bn1b  = (const float*)d_in[4];
    const float* bn1m  = (const float*)d_in[5];
    const float* bn1v  = (const float*)d_in[6];
    const float* w21   = (const float*)d_in[7];
    const float* bn21g = (const float*)d_in[8];
    const float* bn21b = (const float*)d_in[9];
    const float* bn21m = (const float*)d_in[10];
    const float* bn21v = (const float*)d_in[11];
    const float* w22   = (const float*)d_in[12];
    const float* bn22g = (const float*)d_in[13];
    const float* bn22b = (const float*)d_in[14];
    const float* bn22m = (const float*)d_in[15];
    const float* bn22v = (const float*)d_in[16];
    const float* prg   = (const float*)d_in[17];
    const float* prb   = (const float*)d_in[18];
    const float* prz   = (const float*)d_in[19];

    char* ws = (char*)d_ws;
    size_t off = 0;
    auto carve = [&](size_t bytes) -> char* {
        char* p = ws + off;
        off += (bytes + 255) & ~(size_t)255;
        return p;
    };
    signed char* s1   = (signed char*)carve((size_t)NPIX_IN * CCH);
    signed char* s2   = (signed char*)carve((size_t)NPIX * CCH);
    float* out2       = (float*)carve((size_t)NPIX * CCH * 4);
    float* avgp       = (float*)carve((size_t)NPIX * CCH * 4);
    int* Bpack1       = (int*)carve((size_t)KB1 * NT1 * 32 * 8 * 4);
    int* Bpack2       = (int*)carve((size_t)KB2 * NT2 * 32 * 8 * 4);
    float* chanA1     = (float*)carve(256 * 4);
    float* chanB1     = (float*)carve(256 * 4);
    float* chanA2     = (float*)carve(512 * 4);
    float* chanB2     = (float*)carve(512 * 4);
    signed char* zbuf = (signed char*)carve(256);

    (void)hipMemsetAsync(zbuf, 0, 256, stream);
    prep_w3x3<<<256, 256, 0, stream>>>(w1, bn1g, bn1b, bn1m, bn1v, Bpack1, chanA1, chanB1);
    prep_w1x1<<<256, 256, 0, stream>>>(w21, bn21g, bn21b, bn21m, bn21v, Bpack2, chanA2, chanB2, 0);
    prep_w1x1<<<256, 256, 0, stream>>>(w22, bn22g, bn22b, bn22m, bn22v, Bpack2, chanA2, chanB2, 1);
    binarize_x<<<NPIX_IN / 256, 256, 0, stream>>>(x, rsb, s1);
    avgpool_k<<<(BATCH * CCH * HOUT * HOUT) / 256, 256, 0, stream>>>(x, avgp);
    gemm1<<<dim3(NPIX / 128, CCH / 64), 256, 0, stream>>>(
        s1, Bpack1, chanA1, chanB1, avgp, prg, prb, prz, rsb, zbuf, out2, s2);
    gemm2<<<dim3(NPIX / 128, 512 / 64), 256, 0, stream>>>(
        s2, Bpack2, chanA2, chanB2, out2, prg, prb, prz, (float*)d_out);
}
